// VicDyf_86681029968072
// MI455X (gfx1250) — matrix operand order repaired and checked by reading, not compiled, not run on hardware
//
#include <hip/hip_runtime.h>

typedef __attribute__((ext_vector_type(16))) __bf16 v16bf;
typedef __attribute__((ext_vector_type(8)))  __bf16 v8bf;
typedef __attribute__((ext_vector_type(8)))  float  v8f;
typedef __attribute__((ext_vector_type(4)))  float  v4f;
typedef __attribute__((__vector_size__(4 * sizeof(int)))) int v4i;

#define DEV __device__ __forceinline__

// ---- async global->LDS staging (gfx1250), with compile-safe fallback -------
#if defined(__has_builtin)
#if __has_builtin(__builtin_amdgcn_global_load_async_to_lds_b128) && \
    __has_builtin(__builtin_amdgcn_s_wait_asynccnt)
#define USE_ASYNC_LDS 1
#endif
#endif
#ifndef USE_ASYNC_LDS
#define USE_ASYNC_LDS 0
#endif

DEV void stage16(const void* g, void* l){
#if USE_ASYNC_LDS
  __builtin_amdgcn_global_load_async_to_lds_b128(
      (__attribute__((address_space(1))) v4i*)(g),
      (__attribute__((address_space(3))) v4i*)(l), 0, 0);
#else
  *(v4f*)l = *(const v4f*)g;
#endif
}
DEV void wait_async(){
#if USE_ASYNC_LDS
  __builtin_amdgcn_s_wait_asynccnt(0);
#endif
}

DEV float wred16(float x){
#pragma unroll
  for (int m = 1; m < 16; m <<= 1) x += __shfl_xor(x, m, 32);
  return x;
}
DEV float wred32(float x){
#pragma unroll
  for (int m = 1; m < 32; m <<= 1) x += __shfl_xor(x, m, 32);
  return x;
}
DEV float geluf(float x){ return 0.5f * x * (1.0f + erff(x * 0.70710678118654752f)); }
DEV float gelu_grad(float x){
  float Phi = 0.5f * (1.0f + erff(x * 0.70710678118654752f));
  float phi = 0.3989422804014327f * expf(-0.5f * x * x);
  return Phi + x * phi;
}
DEV float softplusf(float x){ return fmaxf(x, 0.f) + log1pf(expf(-fabsf(x))); }
DEV float sigmoidf(float x){ return 1.f / (1.f + expf(-x)); }

DEV v16bf combine8(v8bf lo, v8bf hi){
  return __builtin_shufflevector(lo, hi, 0,1,2,3,4,5,6,7,8,9,10,11,12,13,14,15);
}
DEV v8bf cvt8(v4f a, v4f b){
  v8bf r;
#pragma unroll
  for (int i = 0; i < 4; ++i){ r[i] = (__bf16)a[i]; r[i+4] = (__bf16)b[i]; }
  return r;
}

// ---------------------------------------------------------------------------
// Weight repack into WMMA B-fragment layout.
// fragment (kt,nt): lane L holds column n = nt*16+(L&15), 16 bf16 with k =
// kt*32 + (L>>4)*8 + j (j<8), +16 for j>=8 -> one 32B chunk per lane (1KB/fragment).
// ntMajor=0: frag = kt*NT+nt (per-kt contiguous); ntMajor=1: frag = nt*KT+kt.
// ---------------------------------------------------------------------------
__global__ void k_pack(const float* __restrict__ w, __bf16* __restrict__ dst,
                       int K, int N, int KT, int NT, int ntMajor){
  int tid = blockIdx.x * blockDim.x + threadIdx.x;
  if (tid >= KT * NT * 32) return;
  int lane = tid & 31, frag = tid >> 5;
  int nt, kt;
  if (ntMajor){ kt = frag % KT; nt = frag / KT; }
  else        { nt = frag % NT; kt = frag / NT; }
  int n  = nt * 16 + (lane & 15);
  int kb = kt * 32 + (lane >> 4) * 8;
  __bf16* o = dst + (size_t)frag * 512 + lane * 16;
#pragma unroll
  for (int j = 0; j < 16; ++j){
    int k = kb + (j < 8 ? j : j + 8);
    float v = (k < K && n < N) ? w[(size_t)k * N + n] : 0.f;
    o[j] = (__bf16)v;
  }
}

// Shared epilogue: bias + LayerNorm(100) + exact GELU, store bf16 row-major
// (width 128; cols 100..111 garbage hits zero weights downstream, 112..127
// explicitly zeroed).
DEV void ln_gelu_store(v8f* acc, const float* __restrict__ bias,
                       __bf16* __restrict__ out, int rowbase, int lane){
  int half = lane >> 4, l16 = lane & 15;
#pragma unroll
  for (int t = 0; t < 7; ++t){
    int col = t * 16 + l16;
    float b = (col < 100) ? bias[col] : 0.f;
#pragma unroll
    for (int r = 0; r < 8; ++r) acc[t][r] += b;
  }
#pragma unroll
  for (int r = 0; r < 8; ++r){
    float s1 = 0.f, s2 = 0.f;
#pragma unroll
    for (int t = 0; t < 7; ++t){ float x = acc[t][r]; s1 += x; s2 += x * x; }
    s1 = wred16(s1); s2 = wred16(s2);
    float m   = s1 * 0.01f;
    float inv = rsqrtf(s2 * 0.01f - m * m + 1e-5f);
    int orow = rowbase + half * 8 + r;
#pragma unroll
    for (int t = 0; t < 7; ++t){
      int col = t * 16 + l16;
      out[(size_t)orow * 128 + col] = (__bf16)geluf((acc[t][r] - m) * inv);
    }
    out[(size_t)orow * 128 + 112 + l16] = (__bf16)0.f;
  }
}

// GEMM1: [s,u] (B x 4000) @ ez_w1 -> LN -> GELU -> h1 bf16 (B x 128)
// B fragments double-buffer staged in LDS (7KB per k-tile), async prefetched.
__global__ void __launch_bounds__(256)
k_gemm1(const float* __restrict__ s, const float* __restrict__ u,
        const __bf16* __restrict__ wf, const float* __restrict__ bias,
        __bf16* __restrict__ out){
  __shared__ __align__(32) char bstage[2][7168];
  int tid = threadIdx.x;
  int wave = tid >> 5, lane = tid & 31;
  int rowbase = (blockIdx.x * 8 + wave) * 16;
  int half = lane >> 4, l16 = lane & 15;
  int row = rowbase + l16;

  // prologue: stage kt=0 (7 fragments = 448 x 16B)
  for (int c = tid; c < 448; c += 256)
    stage16((const char*)wf + c * 16, bstage[0] + c * 16);

  v8f acc[7] = {};
  for (int kt = 0; kt < 125; ++kt){
    int buf = kt & 1;
    wait_async();
    __syncthreads();
    if (kt + 1 < 125){
      const char* src = (const char*)wf + (size_t)(kt + 1) * 7168;
      for (int c = tid; c < 448; c += 256)
        stage16(src + c * 16, bstage[buf ^ 1] + c * 16);
    }
    int kb = kt * 32 + half * 8;
    int c0 = kb, c1 = kb + 16;
    const float* p = (c0 < 2000) ? s + (size_t)row * 2000 + c0
                                 : u + (size_t)row * 2000 + (c0 - 2000);
    const float* q = (c1 < 2000) ? s + (size_t)row * 2000 + c1
                                 : u + (size_t)row * 2000 + (c1 - 2000);
    v8bf lo = cvt8(__builtin_nontemporal_load((const v4f*)p),
                   __builtin_nontemporal_load((const v4f*)p + 1));
    v8bf hi = cvt8(__builtin_nontemporal_load((const v4f*)q),
                   __builtin_nontemporal_load((const v4f*)q + 1));
    v16bf a = combine8(lo, hi);
    // hoist all B-fragment LDS loads ahead of the WMMA chain
    v16bf bfr[7];
#pragma unroll
    for (int t = 0; t < 7; ++t)
      bfr[t] = *(const v16bf*)(bstage[buf] + t * 1024 + lane * 32);
#pragma unroll
    for (int t = 0; t < 7; ++t)
      acc[t] = __builtin_amdgcn_wmma_f32_16x16x32_bf16(false, a, false, bfr[t],
                                                       (short)0, acc[t], false, false);
  }
  ln_gelu_store(acc, bias, out, rowbase, lane);
}

// GEMM2: h1 (B x 128 bf16) @ ez_w2 -> LN -> GELU -> h2 bf16 (B x 128)
// Entire 28KB weight staged once in LDS.
__global__ void __launch_bounds__(256)
k_gemm2(const __bf16* __restrict__ h1, const __bf16* __restrict__ wf,
        const float* __restrict__ bias, __bf16* __restrict__ out){
  __shared__ __align__(32) char bst2[28672];
  int tid = threadIdx.x;
  int wave = tid >> 5, lane = tid & 31;
  int rowbase = (blockIdx.x * 8 + wave) * 16;
  int half = lane >> 4, l16 = lane & 15;
  int row = rowbase + l16;

  for (int c = tid; c < 1792; c += 256)
    stage16((const char*)wf + c * 16, bst2 + c * 16);
  wait_async();
  __syncthreads();

  v8f acc[7] = {};
#pragma unroll
  for (int kt = 0; kt < 4; ++kt){
    const __bf16* p = h1 + (size_t)row * 128 + kt * 32 + half * 8;
    v16bf a = combine8(*(const v8bf*)p, *(const v8bf*)(p + 16));
    v16bf bfr[7];
#pragma unroll
    for (int t = 0; t < 7; ++t)
      bfr[t] = *(const v16bf*)(bst2 + (kt * 7 + t) * 1024 + lane * 32);
#pragma unroll
    for (int t = 0; t < 7; ++t)
      acc[t] = __builtin_amdgcn_wmma_f32_16x16x32_bf16(false, a, false, bfr[t],
                                                       (short)0, acc[t], false, false);
  }
  ln_gelu_store(acc, bias, out, rowbase, lane);
}

// Per-row VALU stage: heads -> z, enc_d -> dz, decoder hidden + analytic JVP
// through LayerNorm + exact GELU. One wave per row.
__global__ void __launch_bounds__(256)
k_rowops(const __bf16* __restrict__ h2,
         const float* __restrict__ eps_z, const float* __restrict__ eps_d,
         const float* __restrict__ ez_wmu, const float* __restrict__ ez_bmu,
         const float* __restrict__ ez_wlv, const float* __restrict__ ez_blv,
         const float* __restrict__ ed_w1, const float* __restrict__ ed_b1,
         const float* __restrict__ ed_wmu, const float* __restrict__ ed_bmu,
         const float* __restrict__ ed_wlv, const float* __restrict__ ed_blv,
         const float* __restrict__ dz_w1, const float* __restrict__ dz_b1,
         __bf16* __restrict__ Hb, __bf16* __restrict__ HDb,
         float* __restrict__ o_z, float* __restrict__ o_dz,
         float* __restrict__ o_muz, float* __restrict__ o_scz,
         float* __restrict__ o_mud, float* __restrict__ o_scd){
  __shared__ float lds[8][340];
  int w = threadIdx.x >> 5, lane = threadIdx.x & 31;
  int row = blockIdx.x * 8 + w;
  float* hh  = &lds[w][0];   // 100
  float* zz  = hh + 100;     // 20
  float* dzv = zz + 20;      // 20
  float* aa  = dzv + 20;     // 100
  float* ad  = aa + 100;     // 100

  for (int i = lane; i < 100; i += 32) hh[i] = (float)h2[(size_t)row * 128 + i];
  __syncthreads();

  // enc_z heads
  if (lane < 20){
    float mu = ez_bmu[lane], lv = ez_blv[lane];
    for (int k = 0; k < 100; ++k){
      float h = hh[k];
      mu = fmaf(h, ez_wmu[k * 20 + lane], mu);
      lv = fmaf(h, ez_wlv[k * 20 + lane], lv);
    }
    float sc = softplusf(lv);
    float zv = mu + sc * eps_z[(size_t)row * 20 + lane];
    o_muz[(size_t)row * 20 + lane] = mu;
    o_scz[(size_t)row * 20 + lane] = sc;
    o_z  [(size_t)row * 20 + lane] = zv;
    zz[lane] = zv;
  }
  __syncthreads();

  // enc_d hidden: LinearGELU(z)
  for (int i = lane; i < 100; i += 32){
    float t = ed_b1[i];
    for (int j = 0; j < 20; ++j) t = fmaf(zz[j], ed_w1[j * 100 + i], t);
    aa[i] = t;
  }
  __syncthreads();
  {
    float s1 = 0.f, s2 = 0.f;
    for (int i = lane; i < 100; i += 32){ float x = aa[i]; s1 += x; s2 += x * x; }
    s1 = wred32(s1); s2 = wred32(s2);
    float m = s1 * 0.01f, inv = rsqrtf(s2 * 0.01f - m * m + 1e-5f);
    for (int i = lane; i < 100; i += 32) hh[i] = geluf((aa[i] - m) * inv);
  }
  __syncthreads();

  // enc_d heads
  if (lane < 20){
    float mu = ed_bmu[lane], lv = ed_blv[lane];
    for (int k = 0; k < 100; ++k){
      float h = hh[k];
      mu = fmaf(h, ed_wmu[k * 20 + lane], mu);
      lv = fmaf(h, ed_wlv[k * 20 + lane], lv);
    }
    float sc = softplusf(lv);
    float dv = mu + sc * eps_d[(size_t)row * 20 + lane];
    o_mud[(size_t)row * 20 + lane] = mu;
    o_scd[(size_t)row * 20 + lane] = sc;
    o_dz [(size_t)row * 20 + lane] = dv;
    dzv[lane] = dv;
  }
  __syncthreads();

  // decoder hidden + tangent: a = z@W1+b1, a_dot = dz@W1
  for (int i = lane; i < 100; i += 32){
    float t = dz_b1[i], td = 0.f;
    for (int j = 0; j < 20; ++j){
      float wv = dz_w1[j * 100 + i];
      t  = fmaf(zz[j],  wv, t);
      td = fmaf(dzv[j], wv, td);
    }
    aa[i] = t; ad[i] = td;
  }
  __syncthreads();
  {
    float s1 = 0.f, s2 = 0.f, s3 = 0.f, s4 = 0.f;
    for (int i = lane; i < 100; i += 32){
      float x = aa[i], d = ad[i];
      s1 += x; s2 += x * x; s3 += d; s4 += x * d;
    }
    s1 = wred32(s1); s2 = wred32(s2); s3 = wred32(s3); s4 = wred32(s4);
    float m   = s1 * 0.01f;
    float inv = rsqrtf(s2 * 0.01f - m * m + 1e-5f);
    float md  = s3 * 0.01f;
    float c   = s4 * 0.01f - m * md;   // mean((a-m)*a_dot)
    float cyd = c * inv;               // mean(y*a_dot)
    for (int i = lane; i < 100; i += 32){
      float y  = (aa[i] - m) * inv;
      float yd = inv * (ad[i] - md - y * cyd);
      Hb [(size_t)row * 128 + i] = (__bf16)geluf(y);
      HDb[(size_t)row * 128 + i] = (__bf16)(gelu_grad(y) * yd);
    }
    for (int i = lane; i < 28; i += 32){
      Hb [(size_t)row * 128 + 100 + i] = (__bf16)0.f;
      HDb[(size_t)row * 128 + 100 + i] = (__bf16)0.f;
    }
  }
}

// Decoder GEMM (p and p_dot), two passes over N; weight fragments nt-major,
// 4KB per n-tile double-buffer staged in LDS with async prefetch.
__global__ void __launch_bounds__(256)
k_dec(const __bf16* __restrict__ H, const __bf16* __restrict__ HD,
      const __bf16* __restrict__ wf, const float* __restrict__ b2,
      const float* __restrict__ logbeta, const float* __restrict__ loggamma,
      float* __restrict__ o_shat, float* __restrict__ o_diff,
      float* __restrict__ o_pu){
  __shared__ __align__(32) char dstg[2][4096];
  int tid = threadIdx.x;
  int wave = tid >> 5, lane = tid & 31;
  int rowbase = (blockIdx.x * 8 + wave) * 16;
  int half = lane >> 4, l16 = lane & 15;
  int arow = rowbase + l16;
  v16bf ha[4], hda[4];
#pragma unroll
  for (int kt = 0; kt < 4; ++kt){
    const __bf16* p = H  + (size_t)arow * 128 + kt * 32 + half * 8;
    const __bf16* q = HD + (size_t)arow * 128 + kt * 32 + half * 8;
    ha [kt] = combine8(*(const v8bf*)p, *(const v8bf*)(p + 16));
    hda[kt] = combine8(*(const v8bf*)q, *(const v8bf*)(q + 16));
  }
  float sld[8], sldd[8];
#pragma unroll
  for (int r = 0; r < 8; ++r){ sld[r] = 0.f; sldd[r] = 0.f; }

  // ---- pass 1: accumulate sum(softplus(p)), sum(sigmoid(p)*p_dot) ----
  stage16((const char*)wf + tid * 16, dstg[0] + tid * 16);   // nt=0: 256 chunks
  for (int nt = 0; nt < 125; ++nt){
    int buf = nt & 1;
    wait_async();
    __syncthreads();
    if (nt + 1 < 125)
      stage16((const char*)wf + (size_t)(nt + 1) * 4096 + tid * 16,
              dstg[buf ^ 1] + tid * 16);
    v16bf bfr[4];
#pragma unroll
    for (int kt = 0; kt < 4; ++kt)
      bfr[kt] = *(const v16bf*)(dstg[buf] + kt * 1024 + lane * 32);
    v8f p = {}, pd = {};
#pragma unroll
    for (int kt = 0; kt < 4; ++kt){
      p  = __builtin_amdgcn_wmma_f32_16x16x32_bf16(false, ha [kt], false, bfr[kt], (short)0, p,  false, false);
      pd = __builtin_amdgcn_wmma_f32_16x16x32_bf16(false, hda[kt], false, bfr[kt], (short)0, pd, false, false);
    }
    int col = nt * 16 + l16;
    float bb = b2[col];
#pragma unroll
    for (int r = 0; r < 8; ++r){
      float pv  = p[r] + bb;
      sld [r] += softplusf(pv);
      sldd[r] += sigmoidf(pv) * pd[r];
    }
  }
  float mld[8], mldd[8];
#pragma unroll
  for (int r = 0; r < 8; ++r){
    mld [r] = wred16(sld [r]) * (1.f / 2000.f);
    mldd[r] = wred16(sldd[r]) * (1.f / 2000.f);
  }

  // ---- pass 2: recompute and emit s_hat, diff, pu ----
  __syncthreads();   // all waves done reading pass-1 buffers
  stage16((const char*)wf + tid * 16, dstg[0] + tid * 16);
  for (int nt = 0; nt < 125; ++nt){
    int buf = nt & 1;
    wait_async();
    __syncthreads();
    if (nt + 1 < 125)
      stage16((const char*)wf + (size_t)(nt + 1) * 4096 + tid * 16,
              dstg[buf ^ 1] + tid * 16);
    v16bf bfr[4];
#pragma unroll
    for (int kt = 0; kt < 4; ++kt)
      bfr[kt] = *(const v16bf*)(dstg[buf] + kt * 1024 + lane * 32);
    v8f p = {}, pd = {};
#pragma unroll
    for (int kt = 0; kt < 4; ++kt){
      p  = __builtin_amdgcn_wmma_f32_16x16x32_bf16(false, ha [kt], false, bfr[kt], (short)0, p,  false, false);
      pd = __builtin_amdgcn_wmma_f32_16x16x32_bf16(false, hda[kt], false, bfr[kt], (short)0, pd, false, false);
    }
    int col = nt * 16 + l16;
    float bb    = b2[col];
    float beta  = softplusf(logbeta[col]);   // * DT (=1)
    float gamma = softplusf(loggamma[col]);  // * DT (=1)
    float rb = 1.f / beta;
#pragma unroll
    for (int r = 0; r < 8; ++r){
      float pv   = p[r] + bb;
      float ld   = softplusf(pv);
      float ldd  = sigmoidf(pv) * pd[r];
      float shat = ld / mld[r];
      float diff = 0.01f * ((ldd - shat * mldd[r]) / mld[r]);
      float raw  = (diff + shat * gamma) * rb;
      int orow = rowbase + half * 8 + r;
      size_t o = (size_t)orow * 2000 + col;
      __builtin_nontemporal_store(shat,            &o_shat[o]);
      __builtin_nontemporal_store(diff,            &o_diff[o]);
      __builtin_nontemporal_store(fmaxf(raw, 0.f), &o_pu[o]);   // raw+relu(-raw)==relu(raw)
    }
  }
}

extern "C" void kernel_launch(void* const* d_in, const int* in_sizes, int n_in,
                              void* d_out, int out_size, void* d_ws, size_t ws_size,
                              hipStream_t stream){
  (void)in_sizes; (void)n_in; (void)out_size; (void)ws_size;
  const float* s        = (const float*)d_in[0];
  const float* u        = (const float*)d_in[1];
  const float* eps_z    = (const float*)d_in[2];
  const float* eps_d    = (const float*)d_in[3];
  const float* ez_w1    = (const float*)d_in[4];
  const float* ez_b1    = (const float*)d_in[5];
  const float* ez_w2    = (const float*)d_in[6];
  const float* ez_b2    = (const float*)d_in[7];
  const float* ez_wmu   = (const float*)d_in[8];
  const float* ez_bmu   = (const float*)d_in[9];
  const float* ez_wlv   = (const float*)d_in[10];
  const float* ez_blv   = (const float*)d_in[11];
  const float* ed_w1    = (const float*)d_in[12];
  const float* ed_b1    = (const float*)d_in[13];
  const float* ed_wmu   = (const float*)d_in[14];
  const float* ed_bmu   = (const float*)d_in[15];
  const float* ed_wlv   = (const float*)d_in[16];
  const float* ed_blv   = (const float*)d_in[17];
  const float* dz_w1    = (const float*)d_in[18];
  const float* dz_b1    = (const float*)d_in[19];
  const float* dz_w2    = (const float*)d_in[20];
  const float* dz_b2    = (const float*)d_in[21];
  const float* logbeta  = (const float*)d_in[22];
  const float* loggamma = (const float*)d_in[23];

  const int B = 16384;
  float* out = (float*)d_out;
  float* o_z    = out;
  float* o_dz   = out + (size_t)B * 20;
  float* o_muz  = out + (size_t)B * 40;
  float* o_scz  = out + (size_t)B * 60;
  float* o_mud  = out + (size_t)B * 80;
  float* o_scd  = out + (size_t)B * 100;
  float* o_shat = out + (size_t)B * 120;
  float* o_diff = o_shat + (size_t)B * 2000;
  float* o_pu   = o_diff + (size_t)B * 2000;

  char* ws = (char*)d_ws;
  size_t off = 0;
  auto alloc = [&](size_t bytes) -> char* {
    off = (off + 255) & ~(size_t)255;
    char* p = ws + off; off += bytes; return p;
  };
  __bf16* wez1f = (__bf16*)alloc((size_t)125 * 7 * 512 * 2);
  __bf16* wez2f = (__bf16*)alloc((size_t)4   * 7 * 512 * 2);
  __bf16* wdz2f = (__bf16*)alloc((size_t)4 * 125 * 512 * 2);
  __bf16* h1    = (__bf16*)alloc((size_t)B * 128 * 2);
  __bf16* h2    = (__bf16*)alloc((size_t)B * 128 * 2);
  __bf16* Hb    = (__bf16*)alloc((size_t)B * 128 * 2);
  __bf16* HDb   = (__bf16*)alloc((size_t)B * 128 * 2);

  k_pack<<<(125 * 7 * 32 + 255) / 256, 256, 0, stream>>>(ez_w1, wez1f, 4000, 100, 125, 7, 0);
  k_pack<<<(4 * 7 * 32 + 255) / 256,   256, 0, stream>>>(ez_w2, wez2f, 100, 100, 4, 7, 0);
  k_pack<<<(4 * 125 * 32 + 255) / 256, 256, 0, stream>>>(dz_w2, wdz2f, 100, 2000, 4, 125, 1);

  k_gemm1<<<128, 256, 0, stream>>>(s, u, wez1f, ez_b1, h1);
  k_gemm2<<<128, 256, 0, stream>>>(h1, wez2f, ez_b2, h2);
  k_rowops<<<2048, 256, 0, stream>>>(h2, eps_z, eps_d,
                                     ez_wmu, ez_bmu, ez_wlv, ez_blv,
                                     ed_w1, ed_b1, ed_wmu, ed_bmu, ed_wlv, ed_blv,
                                     dz_w1, dz_b1, Hb, HDb,
                                     o_z, o_dz, o_muz, o_scz, o_mud, o_scd);
  k_dec<<<128, 256, 0, stream>>>(Hb, HDb, wdz2f, dz_b2, logbeta, loggamma,
                                 o_shat, o_diff, o_pu);
}
